// GNNModel_12077448036406
// MI455X (gfx1250) — compile-verified, matrix operand order checked by reading, not run on hardware
//
#include <hip/hip_runtime.h>
#include <hip/hip_bf16.h>

typedef __bf16 bf16_t;
typedef __attribute__((ext_vector_type(16))) __bf16 v16bf;
typedef __attribute__((ext_vector_type(8)))  float  v8f;

#define NEG_SLOPE 0.2f

// ---------------- conversion / fill kernels ----------------
__global__ void k_conv_bf16(const float* __restrict__ in, bf16_t* __restrict__ out, long n) {
  long i = (long)blockIdx.x * blockDim.x + threadIdx.x;
  long stride = (long)gridDim.x * blockDim.x;
  for (; i < n; i += stride) out[i] = (bf16_t)in[i];
}

__global__ void k_relu_to_bf16(const float* __restrict__ in, bf16_t* __restrict__ out, long n) {
  long i = (long)blockIdx.x * blockDim.x + threadIdx.x;
  long stride = (long)gridDim.x * blockDim.x;
  for (; i < n; i += stride) out[i] = (bf16_t)fmaxf(in[i], 0.0f);
}

__global__ void k_fill_u32(unsigned* __restrict__ p, unsigned v, int n) {
  int i = blockIdx.x * blockDim.x + threadIdx.x;
  if (i < n) p[i] = v;
}
__global__ void k_copy_u32(const unsigned* __restrict__ in, unsigned* __restrict__ out, int n) {
  int i = blockIdx.x * blockDim.x + threadIdx.x;
  if (i < n) out[i] = in[i];
}

// ---------------- CSR build: histogram, 3-kernel exclusive scan, scatter ----------------
__global__ void k_hist(const int* __restrict__ ei, unsigned* __restrict__ deg, int E, int ET) {
  int i = blockIdx.x * blockDim.x + threadIdx.x;
  if (i >= ET) return;
  int dst = (i < E) ? ei[E + i] : (i - E);
  atomicAdd(&deg[dst], 1u);
}

__global__ void k_scan1(const unsigned* __restrict__ deg, unsigned* __restrict__ off,
                        unsigned* __restrict__ bsum, int n) {
  __shared__ unsigned s[256];
  const int i = blockIdx.x * 256 + threadIdx.x;
  const unsigned v = (i < n) ? deg[i] : 0u;
  s[threadIdx.x] = v;
  __syncthreads();
  for (int d = 1; d < 256; d <<= 1) {
    unsigned t = (threadIdx.x >= d) ? s[threadIdx.x - d] : 0u;
    __syncthreads();
    s[threadIdx.x] += t;
    __syncthreads();
  }
  if (i < n) off[i] = s[threadIdx.x] - v;            // exclusive
  if (threadIdx.x == 255) bsum[blockIdx.x] = s[255]; // block total
}

__global__ void k_scan2(unsigned* __restrict__ bsum, int nb) {
  __shared__ unsigned s[256];
  const unsigned v = (threadIdx.x < nb) ? bsum[threadIdx.x] : 0u;
  s[threadIdx.x] = v;
  __syncthreads();
  for (int d = 1; d < 256; d <<= 1) {
    unsigned t = (threadIdx.x >= d) ? s[threadIdx.x - d] : 0u;
    __syncthreads();
    s[threadIdx.x] += t;
    __syncthreads();
  }
  if (threadIdx.x < nb) bsum[threadIdx.x] = s[threadIdx.x] - v; // exclusive
}

__global__ void k_scan3(unsigned* __restrict__ off, const unsigned* __restrict__ bsum, int n) {
  int i = blockIdx.x * 256 + threadIdx.x;
  if (i < n) off[i] += bsum[blockIdx.x];
}

__global__ void k_scatter(const int* __restrict__ ei, unsigned* __restrict__ cursor,
                          int* __restrict__ csr_src, int* __restrict__ csr_eid, int E, int ET) {
  int i = blockIdx.x * blockDim.x + threadIdx.x;
  if (i >= ET) return;
  int src, dst;
  if (i < E) { src = ei[i]; dst = ei[E + i]; } else { src = dst = i - E; }
  unsigned pos = atomicAdd(&cursor[dst], 1u);
  csr_src[pos] = src;
  csr_eid[pos] = i;
}

// ---------------- WMMA GEMM: C[M,N] = A[M,K]@B[K,N] (+bias), bf16 in / f32 out ----------------
// 256 threads = 8 wave32s, 4(M) x 2(N); each wave computes 16x32 (2 accumulators, shared A frag).
#define GEMM_BM 64
#define GEMM_BN 64
#define LDS_AS 40   // bf16 elems; 80B row stride -> 16B-aligned fragments, conflict-light
#define LDS_BS 40

__global__ __launch_bounds__(256) void k_gemm_bf16_wmma(
    const bf16_t* __restrict__ A, const bf16_t* __restrict__ B,
    const float* __restrict__ bias, float* __restrict__ C,
    int M, int K, int Nn)
{
  __shared__ bf16_t Asl[GEMM_BM * LDS_AS];
  __shared__ bf16_t Bsl[GEMM_BN * LDS_BS];
  const int tid  = threadIdx.x;
  const int lane = tid & 31;
  const int wave = tid >> 5;
  const int wm   = wave & 3;   // 16-row subtile
  const int wn   = wave >> 2;  // 32-col subtile pair
  const int m0   = blockIdx.x * GEMM_BM;
  const int n0   = blockIdx.y * GEMM_BN;

  // A stage: 64 rows x 32 cols, 8 bf16 (16B) per thread
  const int a_row = tid >> 2;
  const int a_col = (tid & 3) * 8;
  // B stage: 32 k x 64 n, 8 bf16 per thread, transposed into LDS as [n][k]
  const int b_k = tid >> 3;
  const int b_n = (tid & 7) * 8;

  v8f acc0 = {0.f,0.f,0.f,0.f,0.f,0.f,0.f,0.f};
  v8f acc1 = {0.f,0.f,0.f,0.f,0.f,0.f,0.f,0.f};

  const int lm = lane & 15;
  const int kb = (lane < 16) ? 0 : 8;   // per-lane K phase (ISA 7.12.2, 16-bit layouts)
  const int gm_ld = m0 + a_row;

  for (int k0 = 0; k0 < K; k0 += 32) {
    // stage A (zero-pad partial last M block)
    uint4 av;
    if (gm_ld < M) {
      av = *(const uint4*)(A + (size_t)gm_ld * K + k0 + a_col);
    } else {
      av.x = 0u; av.y = 0u; av.z = 0u; av.w = 0u;
    }
    *(uint4*)(&Asl[a_row * LDS_AS + a_col]) = av;

    // stage B transposed (K,N always multiples of 32/64 here)
    {
      uint4 bv = *(const uint4*)(B + (size_t)(k0 + b_k) * Nn + n0 + b_n);
      bf16_t tmp[8];
      *(uint4*)tmp = bv;
#pragma unroll
      for (int q = 0; q < 8; ++q) Bsl[(b_n + q) * LDS_BS + b_k] = tmp[q];
    }

    // prefetch next K tile into caches while this tile computes
    if (k0 + 32 < K) {
      if (gm_ld < M) __builtin_prefetch(A + (size_t)gm_ld * K + k0 + 32 + a_col, 0, 3);
      __builtin_prefetch(B + (size_t)(k0 + 32 + b_k) * Nn + n0 + b_n, 0, 3);
    }
    __syncthreads();

    // fragments per ISA 7.12.2: elems[0..7]=K kb..kb+7, elems[8..15]=K 16+kb..
    const bf16_t* ap  = &Asl[(wm * 16 + lm) * LDS_AS + kb];
    const bf16_t* bp0 = &Bsl[(wn * 32 + lm) * LDS_BS + kb];
    const bf16_t* bp1 = bp0 + 16 * LDS_BS;
    v16bf af, bf0, bf1;
#pragma unroll
    for (int i = 0; i < 8; ++i) {
      af[i]      = ap[i];
      af[8 + i]  = ap[16 + i];
      bf0[i]     = bp0[i];
      bf0[8 + i] = bp0[16 + i];
      bf1[i]     = bp1[i];
      bf1[8 + i] = bp1[16 + i];
    }
    acc0 = __builtin_amdgcn_wmma_f32_16x16x32_bf16(false, af, false, bf0,
                                                   (short)0, acc0, false, false);
    acc1 = __builtin_amdgcn_wmma_f32_16x16x32_bf16(false, af, false, bf1,
                                                   (short)0, acc1, false, false);
    __syncthreads();
  }

  // C layout: lanes 0-15 -> M=r, lanes 16-31 -> M=8+r ; N = lane&15
  const int gn0   = n0 + wn * 32 + lm;
  const int gn1   = gn0 + 16;
  const int mbase = m0 + wm * 16 + ((lane < 16) ? 0 : 8);
  const float bc0 = bias ? bias[gn0] : 0.0f;
  const float bc1 = bias ? bias[gn1] : 0.0f;
#pragma unroll
  for (int r = 0; r < 8; ++r) {
    const int gm = mbase + r;
    if (gm < M) {
      C[(size_t)gm * Nn + gn0] = acc0[r] + bc0;
      C[(size_t)gm * Nn + gn1] = acc1[r] + bc1;
    }
  }
}

// ---------------- attention scores: a_s[n]=h[n]·att_s, a_d[n]=h[n]·att_d ----------------
__global__ void k_attn(const float* __restrict__ h, const float* __restrict__ att_s,
                       const float* __restrict__ att_d, float* __restrict__ as_,
                       float* __restrict__ ad_, int H) {
  __shared__ float ss[256];
  __shared__ float sd[256];
  const int n = blockIdx.x;
  const int c = threadIdx.x;
  const float hv = h[(size_t)n * H + c];
  ss[c] = hv * att_s[c];
  sd[c] = hv * att_d[c];
  __syncthreads();
  for (int off = 128; off > 0; off >>= 1) {
    if (c < off) { ss[c] += ss[c + off]; sd[c] += sd[c + off]; }
    __syncthreads();
  }
  if (c == 0) { as_[n] = ss[0]; ad_[n] = sd[0]; }
}

// ---------------- edge score (no atomics) ----------------
__global__ void k_edge_score(const int* __restrict__ ei, const float* __restrict__ as_,
                             const float* __restrict__ ad_, float* __restrict__ ebuf,
                             int E, int ET) {
  int i = blockIdx.x * blockDim.x + threadIdx.x;
  if (i >= ET) return;
  int src, dst;
  if (i < E) { src = ei[i]; dst = ei[E + i]; } else { src = dst = i - E; }
  float v = as_[src] + ad_[dst];
  ebuf[i] = (v > 0.0f) ? v : NEG_SLOPE * v;
}

// ---------------- per-node softmax stats via CSR (one wave32 per node, shuffle reduce) ----------------
__global__ void k_node_stats(const unsigned* __restrict__ off, const int* __restrict__ csr_eid,
                             const float* __restrict__ ebuf, float* __restrict__ mbuf,
                             float* __restrict__ sbuf, int N) {
  const int widx = (int)((blockIdx.x * (unsigned)blockDim.x + threadIdx.x) >> 5); // node id
  const int lane = threadIdx.x & 31;
  if (widx >= N) return;   // wave-uniform
  const unsigned a = off[widx], b = off[widx + 1];
  float m = -3.4e38f;
  for (unsigned j = a + lane; j < b; j += 32) m = fmaxf(m, ebuf[csr_eid[j]]);
  for (int o = 16; o > 0; o >>= 1) m = fmaxf(m, __shfl_xor(m, o, 32));
  float s = 0.0f;
  for (unsigned j = a + lane; j < b; j += 32) s += __expf(ebuf[csr_eid[j]] - m);
  for (int o = 16; o > 0; o >>= 1) s += __shfl_xor(s, o, 32);
  if (lane == 0) { mbuf[widx] = m; sbuf[widx] = s; }
}

// ---------------- alpha per original edge (no atomics), straight into d_out slice ----------------
__global__ void k_alpha(const int* __restrict__ ei, const float* __restrict__ ebuf,
                        const float* __restrict__ mbuf, const float* __restrict__ sbuf,
                        float* __restrict__ alpha_out, int E, int ET) {
  int i = blockIdx.x * blockDim.x + threadIdx.x;
  if (i >= ET) return;
  int dst = (i < E) ? ei[E + i] : (i - E);
  alpha_out[i] = __expf(ebuf[i] - mbuf[dst]) / sbuf[dst];
}

// ---------------- pull aggregation via CSR: agg[n] = bias + sum alpha_e * h[src] (atomic-free) ----------------
__global__ void k_aggregate(const unsigned* __restrict__ off, const int* __restrict__ csr_src,
                            const int* __restrict__ csr_eid, const float* __restrict__ alpha,
                            const float* __restrict__ h, const float* __restrict__ bias,
                            float* __restrict__ agg, int H) {
  const int n = blockIdx.x;
  const int c = threadIdx.x;            // H threads
  const unsigned a = off[n], b = off[n + 1];
  float acc = bias[c];
  for (unsigned j = a; j < b; ++j) {
    const int src = csr_src[j];
    const float al = alpha[csr_eid[j]];
    acc += al * h[(size_t)src * H + c];
  }
  agg[(size_t)n * H + c] = acc;
}

// ---------------- launch ----------------
extern "C" void kernel_launch(void* const* d_in, const int* in_sizes, int n_in,
                              void* d_out, int out_size, void* d_ws, size_t ws_size,
                              hipStream_t stream) {
  const float* x   = (const float*)d_in[0];
  const int*   ei  = (const int*)d_in[1];
  const float* W1  = (const float*)d_in[2];
  const float* as1 = (const float*)d_in[3];
  const float* ad1 = (const float*)d_in[4];
  const float* b1  = (const float*)d_in[5];
  const float* W2  = (const float*)d_in[6];
  const float* as2 = (const float*)d_in[7];
  const float* ad2 = (const float*)d_in[8];
  const float* b2  = (const float*)d_in[9];
  const float* LW  = (const float*)d_in[10];
  const float* LB  = (const float*)d_in[11];
  float* out = (float*)d_out;

  const int H    = in_sizes[3];            // 256
  const int Fin  = in_sizes[2] / H;        // 512
  const int N    = in_sizes[0] / Fin;      // 50000
  const int E    = in_sizes[1] / 2;        // 800000
  const int OutF = in_sizes[10] / H;       // 64
  const int ET   = E + N;                  // edges + self loops

  // ---- carve workspace ----
  char* ws = (char*)d_ws;
  size_t woff = 0;
  auto carve = [&](size_t bytes) -> char* {
    char* p = ws + woff;
    woff = (woff + bytes + 255) & ~(size_t)255;
    return p;
  };
  bf16_t*   xb   = (bf16_t*)carve((size_t)N * Fin * 2);
  bf16_t*   w1b  = (bf16_t*)carve((size_t)Fin * H * 2);
  bf16_t*   w2b  = (bf16_t*)carve((size_t)H * H * 2);
  bf16_t*   lwb  = (bf16_t*)carve((size_t)H * OutF * 2);
  float*    hbf  = (float*)carve((size_t)N * H * 4);     // GEMM output h (per layer)
  float*    agg  = (float*)carve((size_t)N * H * 4);     // aggregation result
  bf16_t*   hb   = (bf16_t*)carve((size_t)N * H * 2);    // relu(out) in bf16
  float*    asb  = (float*)carve((size_t)N * 4);
  float*    adb  = (float*)carve((size_t)N * 4);
  float*    mbf  = (float*)carve((size_t)N * 4);
  float*    sbf  = (float*)carve((size_t)N * 4);
  float*    eb   = (float*)carve((size_t)ET * 4);
  unsigned* deg  = (unsigned*)carve((size_t)N * 4);
  unsigned* off  = (unsigned*)carve((size_t)(N + 1) * 4);
  unsigned* cur  = (unsigned*)carve((size_t)N * 4);
  unsigned* bsum = (unsigned*)carve((size_t)256 * 4);
  int*      csrc = (int*)carve((size_t)ET * 4);
  int*      ceid = (int*)carve((size_t)ET * 4);

  float* alpha1 = out + (size_t)N * OutF;
  float* alpha2 = alpha1 + ET;

  const int nb_node = (N + 255) / 256;
  const int nb_edge = (ET + 255) / 256;
  const int nb_wave = (N + 7) / 8;   // 8 wave32 per 256-thread block

  // ---- precision conversion (bf16 feeds WMMA; halves A/B bytes) ----
  k_conv_bf16<<<2048, 256, 0, stream>>>(x,  xb,  (long)N * Fin);
  k_conv_bf16<<<256,  256, 0, stream>>>(W1, w1b, (long)Fin * H);
  k_conv_bf16<<<256,  256, 0, stream>>>(W2, w2b, (long)H * H);
  k_conv_bf16<<<64,   256, 0, stream>>>(LW, lwb, (long)H * OutF);

  // ---- build dst-CSR once (shared by both layers) ----
  k_fill_u32<<<nb_node, 256, 0, stream>>>(deg, 0u, N);
  k_hist<<<nb_edge, 256, 0, stream>>>(ei, deg, E, ET);
  k_scan1<<<nb_node, 256, 0, stream>>>(deg, off, bsum, N);
  k_scan2<<<1, 256, 0, stream>>>(bsum, nb_node);
  k_scan3<<<nb_node, 256, 0, stream>>>(off, bsum, N);
  k_fill_u32<<<1, 1, 0, stream>>>(off + N, (unsigned)ET, 1);
  k_copy_u32<<<nb_node, 256, 0, stream>>>(off, cur, N);
  k_scatter<<<nb_edge, 256, 0, stream>>>(ei, cur, csrc, ceid, E, ET);

  // ---- layer 1: h = x @ W1 ----
  dim3 g1((N + GEMM_BM - 1) / GEMM_BM, H / GEMM_BN);
  k_gemm_bf16_wmma<<<g1, 256, 0, stream>>>(xb, w1b, nullptr, hbf, N, Fin, H);

  k_attn<<<N, H, 0, stream>>>(hbf, as1, ad1, asb, adb, H);
  k_edge_score<<<nb_edge, 256, 0, stream>>>(ei, asb, adb, eb, E, ET);
  k_node_stats<<<nb_wave, 256, 0, stream>>>(off, ceid, eb, mbf, sbf, N);
  k_alpha<<<nb_edge, 256, 0, stream>>>(ei, eb, mbf, sbf, alpha1, E, ET);
  k_aggregate<<<N, H, 0, stream>>>(off, csrc, ceid, alpha1, hbf, b1, agg, H);

  // ---- layer 2: h = relu(out1) @ W2 ----
  k_relu_to_bf16<<<2048, 256, 0, stream>>>(agg, hb, (long)N * H);
  dim3 g2((N + GEMM_BM - 1) / GEMM_BM, H / GEMM_BN);
  k_gemm_bf16_wmma<<<g2, 256, 0, stream>>>(hb, w2b, nullptr, hbf, N, H, H);

  k_attn<<<N, H, 0, stream>>>(hbf, as2, ad2, asb, adb, H);
  k_edge_score<<<nb_edge, 256, 0, stream>>>(ei, asb, adb, eb, E, ET);
  k_node_stats<<<nb_wave, 256, 0, stream>>>(off, ceid, eb, mbf, sbf, N);
  k_alpha<<<nb_edge, 256, 0, stream>>>(ei, eb, mbf, sbf, alpha2, E, ET);
  k_aggregate<<<N, H, 0, stream>>>(off, csrc, ceid, alpha2, hbf, b2, agg, H);

  // ---- head: out = relu(out2) @ lin_w + lin_b (f32 directly into d_out) ----
  k_relu_to_bf16<<<2048, 256, 0, stream>>>(agg, hb, (long)N * H);
  dim3 g3((N + GEMM_BM - 1) / GEMM_BM, OutF / GEMM_BN);
  k_gemm_bf16_wmma<<<g3, 256, 0, stream>>>(hb, lwb, LB, out, N, H, OutF);
}
